// Attention_77103252898217
// MI455X (gfx1250) — compile-verified, matrix operand order checked by reading, not run on hardware
//
#include <hip/hip_runtime.h>

// ---------------------------------------------------------------------------
// MI455X (gfx1250) fused attention: bf16 WMMA GEMMs (TDM double-buffered)
// + flash attention (async-to-LDS tile loads).
// ---------------------------------------------------------------------------

typedef __attribute__((ext_vector_type(8)))  __bf16 bf16x8;
typedef __attribute__((ext_vector_type(16))) __bf16 v16bf;
typedef __attribute__((ext_vector_type(8)))  float  v8f;
typedef __attribute__((ext_vector_type(4)))  unsigned uint4v;
typedef __attribute__((ext_vector_type(8)))  unsigned uint8v;

#define B_   2
#define H_   16
#define S_   2048
#define DH_  64
#define E_   1024
#define HD_  1024

__device__ __forceinline__ v8f vzero8() {
  v8f z;
#pragma unroll
  for (int i = 0; i < 8; ++i) z[i] = 0.0f;
  return z;
}

__device__ __forceinline__ v8f wmma_bf16(v16bf a, v16bf b, v8f c) {
  return __builtin_amdgcn_wmma_f32_16x16x32_bf16(
      /*neg_a=*/false, a, /*neg_b=*/false, b,
      /*c_mod=*/(short)0, c, /*reuse_a=*/false, /*reuse_b=*/false);
}

// LDS byte offset of a __shared__ object (low 32 bits of the LDS aperture
// address are the LDS offset per CDNA5 flat-addressing rules).
__device__ __forceinline__ unsigned lds_off_of(const void* p) {
  return (unsigned)(uintptr_t)p;
}

// ---------------------------------------------------------------------------
// Tensor Data Mover: 2D tile load Global -> LDS (D# groups per ISA 8.3/8.4).
// tile_x elements contiguous (x-dim), tile_y rows with stride_elems between
// rows; element size 2 bytes (bf16). Optional LDS padding: after every
// 2^(pad_int_code+1) DWORDs stored, skip (pad_amt_code+1) DWORDs.
// ---------------------------------------------------------------------------
__device__ __forceinline__ void tdm_load_2d(unsigned lds_addr, const void* gptr,
                                            unsigned tile_x, unsigned tile_y,
                                            unsigned stride_elems,
                                            unsigned pad_int_code,
                                            unsigned pad_amt_code,
                                            unsigned pad_en) {
  unsigned long long ga = (unsigned long long)(uintptr_t)gptr;
  uint4v g0;
  g0[0] = 1u;                                   // count=1, user descriptor
  g0[1] = lds_addr;                             // lds_addr
  g0[2] = (unsigned)ga;                         // global_addr[31:0]
  g0[3] = (unsigned)((ga >> 32) & 0x1FFFFFFull) // global_addr[56:32]
          | (2u << 30);                         // type = 2 ("image")
  uint8v g1;
  g1[0] = (1u << 16)                            // data_size = 1 (2 bytes)
          | (pad_en << 20) | (pad_int_code << 22) | (pad_amt_code << 25);
  const unsigned td0 = 1u << 20, td1 = 1u << 20;  // generous dims (no OOB)
  g1[1] = (td0 & 0xFFFFu) << 16;
  g1[2] = (td0 >> 16) | ((td1 & 0xFFFFu) << 16);
  g1[3] = (td1 >> 16) | (tile_x << 16);
  g1[4] = tile_y & 0xFFFFu;                     // tile_dim1; tile_dim2 = 0
  g1[5] = stride_elems;                         // tensor_dim0_stride[31:0]
  g1[6] = 0u;
  g1[7] = 0u;
  asm volatile("tensor_load_to_lds %0, %1" :: "s"(g0), "s"(g1) : "memory");
}

// Per-lane async copy of 16 bytes Global -> LDS (ASYNCcnt-tracked).
__device__ __forceinline__ void async_ld16(unsigned lds_addr, const void* g) {
  asm volatile("global_load_async_to_lds_b128 %0, %1, off"
               :: "v"(lds_addr), "v"((unsigned long long)(uintptr_t)g)
               : "memory");
}

__device__ __forceinline__ void wait_asynccnt0() {
  asm volatile("s_wait_asynccnt 0x0" ::: "memory");
}

// Load a 16x32 A-layout fragment (also used for B via [n][k]-stored operand).
// Lane L<16 : row row0+L,    K = kbase + {0..7, 16..23}
// Lane L>=16: row row0+L-16, K = kbase + {8..15, 24..31}
__device__ __forceinline__ v16bf load_frag(const __bf16* base, int stride,
                                           int row0, int kbase, int lane) {
  const int half = lane >> 4;
  const int l = lane & 15;
  const __bf16* p = base + (size_t)(row0 + l) * stride + kbase;
  bf16x8 lo = *(const bf16x8*)(p + (half ? 8 : 0));
  bf16x8 hi = *(const bf16x8*)(p + (half ? 24 : 16));
  return __builtin_shufflevector(lo, hi, 0, 1, 2, 3, 4, 5, 6, 7,
                                 8, 9, 10, 11, 12, 13, 14, 15);
}

// ---------------------------------------------------------------------------
// fp32 -> bf16 conversion
// ---------------------------------------------------------------------------
__global__ void f32_to_bf16(const float* __restrict__ in,
                            __bf16* __restrict__ out, int n) {
  int i = blockIdx.x * blockDim.x + threadIdx.x;
  if (i < n) out[i] = (__bf16)in[i];
}

// ---------------------------------------------------------------------------
// 128x128-tile GEMM: C[m][n] = sum_k A[m][k] * Wn[n][k]   (K = 1024)
// A/B tiles (128x32 bf16) streamed by the Tensor Data Mover, double-buffered.
// MODE 0: write bf16 into [B][H][S][Dh] layout (QKV projections)
// MODE 1: write fp32 row-major [M][N]        (output projection)
// ---------------------------------------------------------------------------
template <int MODE>
__global__ __launch_bounds__(256) void gemm128(const __bf16* __restrict__ A,
                                               const __bf16* __restrict__ Wn,
                                               void* __restrict__ outp) {
  // Row stride 40 elements = 64B data + 16B pad (TDM pad: 16 DWORDs -> +4).
  __shared__ __align__(16) __bf16 As[2][128][40];
  __shared__ __align__(16) __bf16 Bs[2][128][40];

  const int tid = threadIdx.x;
  const int lane = tid & 31;
  const int w = tid >> 5;            // 8 waves
  const int wr = w >> 2;             // 0..1  -> 64-row strip
  const int wc = w & 3;              // 0..3  -> 32-col strip
  const int m0 = blockIdx.x * 128;
  const int n0 = blockIdx.y * 128;

  v8f acc[4][2];
#pragma unroll
  for (int i = 0; i < 4; ++i)
#pragma unroll
    for (int j = 0; j < 2; ++j) acc[i][j] = vzero8();

  const __bf16* a_base = A + (size_t)m0 * E_;
  const __bf16* b_base = Wn + (size_t)n0 * E_;

  if (w == 0) {  // prologue: DMA first K-tile pair into buffer 0
    tdm_load_2d(lds_off_of(&As[0][0][0]), a_base, 32, 128, E_, 3, 3, 1);
    tdm_load_2d(lds_off_of(&Bs[0][0][0]), b_base, 32, 128, E_, 3, 3, 1);
  }

  const int NK = E_ / 32;            // 32 K-steps
  for (int ks = 0; ks < NK; ++ks) {
    const int cur = ks & 1;
    __syncthreads();                 // prior reads of next-buffer done
    if (w == 0) {
      if (ks + 1 < NK) {
        tdm_load_2d(lds_off_of(&As[1 - cur][0][0]),
                    a_base + (ks + 1) * 32, 32, 128, E_, 3, 3, 1);
        tdm_load_2d(lds_off_of(&Bs[1 - cur][0][0]),
                    b_base + (ks + 1) * 32, 32, 128, E_, 3, 3, 1);
        __builtin_amdgcn_s_wait_tensorcnt(2);  // current pair landed
      } else {
        __builtin_amdgcn_s_wait_tensorcnt(0);
      }
    }
    __syncthreads();                 // tile visible to all waves

    v16bf af[4], bfb[2];
#pragma unroll
    for (int mt = 0; mt < 4; ++mt)
      af[mt] = load_frag(&As[cur][0][0], 40, 64 * wr + 16 * mt, 0, lane);
#pragma unroll
    for (int nt = 0; nt < 2; ++nt)
      bfb[nt] = load_frag(&Bs[cur][0][0], 40, 32 * wc + 16 * nt, 0, lane);
#pragma unroll
    for (int mt = 0; mt < 4; ++mt)
#pragma unroll
      for (int nt = 0; nt < 2; ++nt)
        acc[mt][nt] = wmma_bf16(af[mt], bfb[nt], acc[mt][nt]);
  }

  // Epilogue: C layout -> lane (half,ln): row = base + r + 8*half, col = ln
  const int half = lane >> 4;
  const int ln = lane & 15;
#pragma unroll
  for (int mt = 0; mt < 4; ++mt)
#pragma unroll
    for (int nt = 0; nt < 2; ++nt)
#pragma unroll
      for (int rr = 0; rr < 8; ++rr) {
        int m = m0 + 64 * wr + 16 * mt + rr + 8 * half;
        int n = n0 + 32 * wc + 16 * nt + ln;
        float v = acc[mt][nt][rr];
        if (MODE == 0) {
          int b = m >> 11, s = m & (S_ - 1);
          int h = n >> 6, d = n & (DH_ - 1);
          ((__bf16*)outp)[(((size_t)(b * H_ + h)) * S_ + s) * DH_ + d] =
              (__bf16)v;
        } else {
          ((float*)outp)[(size_t)m * HD_ + n] = v;
        }
      }
}

// ---------------------------------------------------------------------------
// RoPE (in place on bf16 [B][H][S][Dh]); one thread per (b,h,s,pair)
// ---------------------------------------------------------------------------
__global__ void rope_kernel(__bf16* __restrict__ t) {
  size_t idx = (size_t)blockIdx.x * blockDim.x + threadIdx.x;
  const size_t total = (size_t)B_ * H_ * S_ * (DH_ / 2);
  if (idx >= total) return;
  int i = (int)(idx & 31);                 // pair index 0..31
  size_t rest = idx >> 5;                  // b*H*S + h*S + s
  int s = (int)(rest & (S_ - 1));
  float inv_freq = __expf(-((float)(2 * i) * (1.0f / (float)DH_)) *
                          9.210340371976184f);  // ln(10000)
  float ang = (float)s * inv_freq;
  float sn = sinf(ang), cs = cosf(ang);
  size_t p0 = rest * DH_ + 2 * i;
  float x1 = (float)t[p0], x2 = (float)t[p0 + 1];
  t[p0] = (__bf16)(x1 * cs - x2 * sn);
  t[p0 + 1] = (__bf16)(x1 * sn + x2 * cs);
}

// ---------------------------------------------------------------------------
// Flash attention: one block = (b,h, 128 query rows); 8 waves x 16 rows.
// Q/K tiles fetched with global_load_async_to_lds_b128; key tiles of 64,
// online softmax, WMMA for QK^T and PV.
// ---------------------------------------------------------------------------
__global__ __launch_bounds__(256) void flash_attn(
    const __bf16* __restrict__ Q, const __bf16* __restrict__ K,
    const __bf16* __restrict__ V, __bf16* __restrict__ O) {
  __shared__ __align__(16) __bf16 Qs[128][72];
  __shared__ __align__(16) __bf16 Ks[64][72];
  __shared__ __align__(16) __bf16 Vst[64][72];   // V transposed: [d][key]
  __shared__ __align__(16) __bf16 Ps[8][16][72]; // per-wave P tile

  const int tid = threadIdx.x;
  const int lane = tid & 31;
  const int w = tid >> 5;
  const int half = lane >> 4;
  const int ln = lane & 15;
  const int qt = blockIdx.x;                 // 0..15
  const int bh = blockIdx.y;                 // 0..31
  const int q0 = qt * 128;
  const size_t base = (size_t)bh * S_ * DH_;
  const int qrow0 = q0 + w * 16;
  const float scale = 0.03125f;              // 1/sqrt(1024)

  // Async-load Q tile 128x64 into padded LDS rows.
  {
    int rr = tid >> 1;
    int cc = (tid & 1) * 32;
    const __bf16* src = Q + base + (size_t)(q0 + rr) * DH_ + cc;
#pragma unroll
    for (int i = 0; i < 4; ++i)
      async_ld16(lds_off_of(&Qs[rr][cc + 8 * i]), src + 8 * i);
  }

  float mrow[8], lrow[8];
  v8f oacc[4];
#pragma unroll
  for (int i = 0; i < 8; ++i) { mrow[i] = -1e30f; lrow[i] = 0.0f; }
#pragma unroll
  for (int i = 0; i < 4; ++i) oacc[i] = vzero8();

  const int jmax = (q0 + 127) >> 6;
  for (int j = 0; j <= jmax; ++j) {
    __syncthreads();                         // prior tile reads complete
    {
      int rr = tid >> 2;                     // key row 0..63
      int cc = (tid & 3) * 16;               // d col
      const __bf16* ksrc = K + base + (size_t)(j * 64 + rr) * DH_ + cc;
      async_ld16(lds_off_of(&Ks[rr][cc]), ksrc);
      async_ld16(lds_off_of(&Ks[rr][cc + 8]), ksrc + 8);
      // V needs a transpose -> stage through VGPRs and scatter.
      const bf16x8* vsrc =
          (const bf16x8*)(V + base + (size_t)(j * 64 + rr) * DH_ + cc);
      bf16x8 v0 = vsrc[0], v1 = vsrc[1];
#pragma unroll
      for (int i = 0; i < 8; ++i) {
        Vst[cc + i][rr] = v0[i];
        Vst[cc + 8 + i][rr] = v1[i];
      }
    }
    wait_asynccnt0();                        // this wave's Q/K copies landed
    __syncthreads();                         // all waves' tiles visible

    // S = Q K^T  (16 rows x 64 keys per wave)
    v8f sacc[4];
#pragma unroll
    for (int nt = 0; nt < 4; ++nt) sacc[nt] = vzero8();
#pragma unroll
    for (int kk = 0; kk < 2; ++kk) {
      v16bf aq = load_frag(&Qs[0][0], 72, w * 16, kk * 32, lane);
#pragma unroll
      for (int nt = 0; nt < 4; ++nt) {
        v16bf bk = load_frag(&Ks[0][0], 72, nt * 16, kk * 32, lane);
        sacc[nt] = wmma_bf16(aq, bk, sacc[nt]);
      }
    }

    // scale + causal mask + online softmax
    const int keybase = j * 64;
    float sval[4][8];
    float lm[8];
#pragma unroll
    for (int rr = 0; rr < 8; ++rr) lm[rr] = -1e30f;
#pragma unroll
    for (int nt = 0; nt < 4; ++nt) {
      int col = keybase + nt * 16 + ln;
#pragma unroll
      for (int rr = 0; rr < 8; ++rr) {
        int rowg = qrow0 + rr + 8 * half;
        float sv = sacc[nt][rr] * scale;
        sv = (col <= rowg) ? sv : -1e30f;
        sval[nt][rr] = sv;
        lm[rr] = fmaxf(lm[rr], sv);
      }
    }
#pragma unroll
    for (int off = 1; off < 16; off <<= 1)
#pragma unroll
      for (int rr = 0; rr < 8; ++rr)
        lm[rr] = fmaxf(lm[rr], __shfl_xor(lm[rr], off, 32));

    float alpha[8], rsum[8];
#pragma unroll
    for (int rr = 0; rr < 8; ++rr) {
      float mn = fmaxf(mrow[rr], lm[rr]);
      alpha[rr] = __expf(mrow[rr] - mn);
      mrow[rr] = mn;
      rsum[rr] = 0.0f;
    }
#pragma unroll
    for (int nt = 0; nt < 4; ++nt)
#pragma unroll
      for (int rr = 0; rr < 8; ++rr) {
        float p = __expf(sval[nt][rr] - mrow[rr]);
        rsum[rr] += p;
        Ps[w][rr + 8 * half][nt * 16 + ln] = (__bf16)p;
      }
#pragma unroll
    for (int off = 1; off < 16; off <<= 1)
#pragma unroll
      for (int rr = 0; rr < 8; ++rr) rsum[rr] += __shfl_xor(rsum[rr], off, 32);
#pragma unroll
    for (int rr = 0; rr < 8; ++rr) lrow[rr] = lrow[rr] * alpha[rr] + rsum[rr];
#pragma unroll
    for (int dt = 0; dt < 4; ++dt)
#pragma unroll
      for (int rr = 0; rr < 8; ++rr) oacc[dt][rr] *= alpha[rr];

    __syncthreads();  // make Ps stores visible (and quiesce DS counters)

    // O += P V  (P: 16x64, V: 64x64 via transposed LDS tile)
#pragma unroll
    for (int kk = 0; kk < 2; ++kk) {
      v16bf ap = load_frag(&Ps[w][0][0], 72, 0, kk * 32, lane);
#pragma unroll
      for (int dt = 0; dt < 4; ++dt) {
        v16bf bv = load_frag(&Vst[0][0], 72, dt * 16, kk * 32, lane);
        oacc[dt] = wmma_bf16(ap, bv, oacc[dt]);
      }
    }
  }

  // Normalize and write out as [B][S][HD] bf16
  const int b = bh >> 4;
  const int h = bh & 15;
#pragma unroll
  for (int dt = 0; dt < 4; ++dt)
#pragma unroll
    for (int rr = 0; rr < 8; ++rr) {
      int srow = qrow0 + rr + 8 * half;
      int d = dt * 16 + ln;
      float ov = oacc[dt][rr] / lrow[rr];
      O[((size_t)(b * S_ + srow)) * HD_ + h * DH_ + d] = (__bf16)ov;
    }
}

// ---------------------------------------------------------------------------
// Host launch
// ---------------------------------------------------------------------------
extern "C" void kernel_launch(void* const* d_in, const int* in_sizes, int n_in,
                              void* d_out, int out_size, void* d_ws,
                              size_t ws_size, hipStream_t stream) {
  (void)in_sizes; (void)n_in; (void)out_size; (void)ws_size;
  const float* x = (const float*)d_in[0];
  const float* Wq = (const float*)d_in[1];
  const float* Wk = (const float*)d_in[2];
  const float* Wv = (const float*)d_in[3];
  const float* Wo = (const float*)d_in[4];

  char* ws = (char*)d_ws;
  const size_t MB = 1024 * 1024;
  __bf16* xb = (__bf16*)(ws + 0);         // 8 MB
  __bf16* wqb = (__bf16*)(ws + 8 * MB);   // 2 MB each
  __bf16* wkb = (__bf16*)(ws + 10 * MB);
  __bf16* wvb = (__bf16*)(ws + 12 * MB);
  __bf16* wob = (__bf16*)(ws + 14 * MB);
  __bf16* qb = (__bf16*)(ws + 16 * MB);   // 8 MB each
  __bf16* kb = (__bf16*)(ws + 24 * MB);
  __bf16* vb = (__bf16*)(ws + 32 * MB);
  __bf16* ab = (__bf16*)(ws + 40 * MB);

  const int nx = B_ * S_ * E_;
  const int nw = HD_ * E_;
  f32_to_bf16<<<(nx + 255) / 256, 256, 0, stream>>>(x, xb, nx);
  f32_to_bf16<<<(nw + 255) / 256, 256, 0, stream>>>(Wq, wqb, nw);
  f32_to_bf16<<<(nw + 255) / 256, 256, 0, stream>>>(Wk, wkb, nw);
  f32_to_bf16<<<(nw + 255) / 256, 256, 0, stream>>>(Wv, wvb, nw);
  f32_to_bf16<<<(nw + 255) / 256, 256, 0, stream>>>(Wo, wob, nw);

  dim3 ggrid((B_ * S_) / 128, HD_ / 128); // 32 x 8
  gemm128<0><<<ggrid, 256, 0, stream>>>(xb, wqb, qb);
  gemm128<0><<<ggrid, 256, 0, stream>>>(xb, wkb, kb);
  gemm128<0><<<ggrid, 256, 0, stream>>>(xb, wvb, vb);

  const int nrope = B_ * H_ * S_ * (DH_ / 2);
  rope_kernel<<<(nrope + 255) / 256, 256, 0, stream>>>(qb);
  rope_kernel<<<(nrope + 255) / 256, 256, 0, stream>>>(kb);

  dim3 fgrid(S_ / 128, B_ * H_);          // 16 x 32
  flash_attn<<<fgrid, 256, 0, stream>>>(qb, kb, vb, ab);

  gemm128<1><<<ggrid, 256, 0, stream>>>(ab, wob, d_out);
}
